// MultiheadFlashDiff2_4922032521593
// MI455X (gfx1250) — compile-verified, matrix operand order checked by reading, not run on hardware
//
#include <hip/hip_runtime.h>
#include <hip/hip_bf16.h>
#include <cmath>

// ---------------- problem constants ----------------
#define B_    2
#define S_    2048
#define E_    2048
#define H_    16
#define HKV_  8
#define D_    64
#define DV_   128          // 2*D
#define SCALE_ 0.125f      // D^-0.5
#define NEGINF_ (-1e30f)

typedef __attribute__((ext_vector_type(16))) __bf16 v16bf;
typedef __attribute__((ext_vector_type(8)))  float  v8f;

// ---- CDNA5 TDM availability (device pass only; host pass falls back) ----
#if defined(__has_builtin)
# if __has_builtin(__builtin_amdgcn_tensor_load_to_lds) && __has_builtin(__builtin_amdgcn_s_wait_tensorcnt)
#  define USE_TDM 1
# endif
#endif
#ifndef USE_TDM
# define USE_TDM 0
#endif

// ---------------- helpers ----------------
static __device__ __forceinline__ unsigned short f2bf(float f) {
    unsigned int u = __float_as_uint(f);
    u += 0x7fffu + ((u >> 16) & 1u);        // round-to-nearest-even
    return (unsigned short)(u >> 16);
}

// Build one 16x32 bf16 WMMA operand fragment for this lane.
// Per ISA layout: lane row = l&15, elements are two contiguous 8-elem chunks
// at K offsets {half*8} and {16+half*8}.  p points at the row start.
static __device__ __forceinline__ v16bf frag16(const unsigned short* p, int half) {
    union { uint4 u[2]; v16bf v; } t;
    t.u[0] = *reinterpret_cast<const uint4*>(p + half * 8);
    t.u[1] = *reinterpret_cast<const uint4*>(p + 16 + half * 8);
    return t.v;
}

static __device__ __forceinline__ v8f wmma_bf16(v16bf a, v16bf b, v8f c) {
    return __builtin_amdgcn_wmma_f32_16x16x32_bf16(false, a, false, b, (short)0, c, false, false);
}

#if USE_TDM
typedef __attribute__((ext_vector_type(4))) unsigned int t4u;
typedef __attribute__((ext_vector_type(8))) int t8i;
typedef __attribute__((ext_vector_type(4))) int t4i;

// Generic pointer to LDS: low 32 bits are the wave-relative LDS byte offset.
static __device__ __forceinline__ unsigned int lds_off(const void* p) {
    return (unsigned int)(uintptr_t)p;
}

// TDM 2D tile load: rows x cols bf16 elements, global row stride = stride_elems,
// LDS gets padded rows (pad config in flags).
static __device__ __forceinline__ void tdm_load_2d(
    unsigned int lds_addr, const unsigned short* g,
    int rows, int cols, int stride_elems, unsigned int flags)
{
    unsigned long long ga = (unsigned long long)(uintptr_t)g;
    t4u g0;
    g0.x = 1u;                                   // count=1, user descriptor
    g0.y = lds_addr;                             // D#.lds_addr
    g0.z = (unsigned int)ga;                     // global_addr[31:0]
    g0.w = (unsigned int)((ga >> 32) & 0x01FFFFFFu) | (2u << 30);  // addr[56:32], type=2
    t8i g1;
    g1[0] = (int)flags;                          // data_size=2B, pad config
    g1[1] = 0;                                   // atomic barrier addr, tensor_dim0 lo16 (=0)
    g1[2] = 0x4000;                              // tensor_dim0 = 1<<30 (hi16); tensor_dim1 lo16=0
    g1[3] = (int)(0x4000u | ((unsigned)cols << 16));  // tensor_dim1 hi16; tile_dim0 = cols
    g1[4] = rows;                                // tile_dim1 = rows, tile_dim2 = 0
    g1[5] = stride_elems;                        // tensor_dim0_stride[31:0]
    g1[6] = 0;                                   // stride[47:32], dim1_stride lo
    g1[7] = 0;
    t4i gz = {0, 0, 0, 0};
#if defined(__clang_major__) && (__clang_major__ >= 23)
    t8i gz8 = {0, 0, 0, 0, 0, 0, 0, 0};
    __builtin_amdgcn_tensor_load_to_lds(g0, g1, gz, gz, gz8, 0);
#else
    __builtin_amdgcn_tensor_load_to_lds(g0, g1, gz, gz, 0);
#endif
}
#endif  // USE_TDM

// ---------------- f32 -> bf16 convert ----------------
__global__ __launch_bounds__(256)
void cvt_kernel(const float* __restrict__ src, unsigned short* __restrict__ dst, int n) {
    int i = blockIdx.x * 256 + threadIdx.x;
    if (i < n) dst[i] = f2bf(src[i]);
}

// ---------------- bf16 GEMM:  C[M,N] = A[M,K] * Bw[N,K]^T ----------------
#define BM 128
#define BN 128
#define BK 64
#define BKP 72   // padded LDS row stride (elements): 64 + 8
// data_size=1 (2B) | pad_enable | pad_interval=4 (32 DWORDs = 128B row) | pad_amount=3 (4 DWORDs = 16B)
#define TDM_FLAGS ((1u << 16) | (1u << 20) | (4u << 22) | (3u << 25))

__global__ __launch_bounds__(256)
void gemm_bf16_kernel(const unsigned short* __restrict__ A,
                      const unsigned short* __restrict__ Bw,
                      float* __restrict__ C,
                      int M, int N, int K) {
#if USE_TDM
    __shared__ __align__(16) unsigned short As[2][BM * BKP];
    __shared__ __align__(16) unsigned short Bs[2][BM * BKP];
#else
    __shared__ __align__(16) unsigned short As[1][BM * BKP];
    __shared__ __align__(16) unsigned short Bs[1][BM * BKP];
#endif
    const int tid  = threadIdx.x;
    const int lane = tid & 31;
    const int w    = tid >> 5;
    const int half = lane >> 4;
    const int l16  = lane & 15;
    const int wrow = w >> 2;        // 0..1 -> 64 rows each
    const int wcol = w & 3;         // 0..3 -> 32 cols each
    const int m0 = blockIdx.y * BM;
    const int n0 = blockIdx.x * BN;
    const int nk = K / BK;

    v8f acc[4][2] = {};

    // Hoist ALL fragment loads (distinct registers) before the 16 WMMAs so the
    // scheduler can issue 24 ds_load_b128 up front and stage the dscnt waits.
    auto compute = [&](int cur) {
        v16bf af[2][4], bfr[2][2];
#pragma unroll
        for (int ks = 0; ks < 2; ++ks) {
#pragma unroll
            for (int mi = 0; mi < 4; ++mi)
                af[ks][mi] = frag16(&As[cur][(wrow * 64 + mi * 16 + l16) * BKP + ks * 32], half);
#pragma unroll
            for (int ni = 0; ni < 2; ++ni)
                bfr[ks][ni] = frag16(&Bs[cur][(wcol * 32 + ni * 16 + l16) * BKP + ks * 32], half);
        }
#pragma unroll
        for (int ks = 0; ks < 2; ++ks)
#pragma unroll
            for (int mi = 0; mi < 4; ++mi)
#pragma unroll
                for (int ni = 0; ni < 2; ++ni)
                    acc[mi][ni] = wmma_bf16(af[ks][mi], bfr[ks][ni], acc[mi][ni]);
    };

#if USE_TDM
    // Each of the 8 waves DMAs its own 16-row slice of the A and B tiles.
    const int sr = w * 16;
    tdm_load_2d(lds_off(&As[0][sr * BKP]), &A[(size_t)(m0 + sr) * K], 16, BK, K, TDM_FLAGS);
    tdm_load_2d(lds_off(&Bs[0][sr * BKP]), &Bw[(size_t)(n0 + sr) * K], 16, BK, K, TDM_FLAGS);
    __builtin_amdgcn_s_wait_tensorcnt(0);
    __syncthreads();
    for (int kt = 0; kt < nk; ++kt) {
        const int cur = kt & 1;
        if (kt + 1 < nk) {   // prefetch next tiles into the other buffer (async DMA)
            const int k0n = (kt + 1) * BK;
            tdm_load_2d(lds_off(&As[cur ^ 1][sr * BKP]), &A[(size_t)(m0 + sr) * K + k0n], 16, BK, K, TDM_FLAGS);
            tdm_load_2d(lds_off(&Bs[cur ^ 1][sr * BKP]), &Bw[(size_t)(n0 + sr) * K + k0n], 16, BK, K, TDM_FLAGS);
        }
        compute(cur);
        if (kt + 1 < nk) __builtin_amdgcn_s_wait_tensorcnt(0);
        __syncthreads();
    }
#else
    for (int kt = 0; kt < nk; ++kt) {
        const int k0 = kt * BK;
        __syncthreads();
#pragma unroll
        for (int i = 0; i < 4; ++i) {
            int c   = tid + i * 256;          // 0..1023
            int row = c >> 3;
            int cc  = (c & 7) * 8;
            *reinterpret_cast<uint4*>(&As[0][row * BKP + cc]) =
                *reinterpret_cast<const uint4*>(&A[(size_t)(m0 + row) * K + k0 + cc]);
            *reinterpret_cast<uint4*>(&Bs[0][row * BKP + cc]) =
                *reinterpret_cast<const uint4*>(&Bw[(size_t)(n0 + row) * K + k0 + cc]);
        }
        __syncthreads();
        compute(0);
    }
#endif

#pragma unroll
    for (int mi = 0; mi < 4; ++mi)
#pragma unroll
        for (int ni = 0; ni < 2; ++ni)
#pragma unroll
            for (int r = 0; r < 8; ++r) {
                int m = m0 + wrow * 64 + mi * 16 + r + half * 8;
                int n = n0 + wcol * 32 + ni * 16 + l16;
                C[(size_t)m * N + n] = acc[mi][ni][r];
            }
}

// ---------------- interleaved rotary + bf16, relayout to [b][h][comp][S][D] ----
__global__ __launch_bounds__(256)
void rope_kernel(const float* __restrict__ src,   // (B,S,C,D) f32
                 const float* __restrict__ cosT,  // (S, D/2)
                 const float* __restrict__ sinT,
                 unsigned short* __restrict__ dst, // [b][c>>1][c&1][S][D] bf16
                 int C, int total) {
    int gid = blockIdx.x * 256 + threadIdx.x;
    if (gid >= total) return;
    int d2 = gid & 31;
    int t  = gid >> 5;
    int c  = t % C;  t /= C;
    int s  = t % S_;
    int b  = t / S_;
    int h = c >> 1, comp = c & 1;
    const float* ip = src + ((size_t)(b * S_ + s) * C + c) * D_ + d2 * 2;
    float x1 = ip[0], x2 = ip[1];
    float cv = cosT[s * 32 + d2], sv = sinT[s * 32 + d2];
    int Hh = C >> 1;
    unsigned short* op = dst + (((size_t)(b * Hh + h) * 2 + comp) * S_ + s) * D_ + d2 * 2;
    op[0] = f2bf(x1 * cv - x2 * sv);
    op[1] = f2bf(x1 * sv + x2 * cv);
}

// ---------------- V: (B,S,HKV*2D) f32  ->  vT [b][hk][dv][S] bf16 ----------------
__global__ __launch_bounds__(256)
void vtrans_kernel(const float* __restrict__ Vf, unsigned short* __restrict__ vT, int total) {
    int gid = blockIdx.x * 256 + threadIdx.x;
    if (gid >= total) return;
    int c = gid & 1023;
    int t = gid >> 10;
    int s = t % S_, b = t / S_;
    int hk = c >> 7, dv = c & 127;
    vT[((size_t)(b * HKV_ + hk) * DV_ + dv) * S_ + s] = f2bf(Vf[gid]);
}

// ---------------- lambda scalar ----------------
__global__ void lam_kernel(const float* lq1, const float* lk1,
                           const float* lq2, const float* lk2,
                           float lambda_init, float* out) {
    __shared__ float r1[64], r2[64];
    int i = threadIdx.x;
    r1[i] = lq1[i] * lk1[i];
    r2[i] = lq2[i] * lk2[i];
    __syncthreads();
    for (int off = 32; off > 0; off >>= 1) {
        if (i < off) { r1[i] += r1[i + off]; r2[i] += r2[i + off]; }
        __syncthreads();
    }
    if (i == 0) out[0] = expf(r1[0]) - expf(r2[0]) + lambda_init;
}

// ---------------- flash attention (one comp), causal, Dq=64, Dv=128 ----------------
#define PKP 40   // padded stride for the P tile in LDS (32 + 8)

__global__ __launch_bounds__(256)
void attn_kernel(const unsigned short* __restrict__ qb,  // [b][h][comp][S][64]
                 const unsigned short* __restrict__ kb,  // [b][hk][comp][S][64]
                 const unsigned short* __restrict__ vT,  // [b][hk][dv][S]
                 float* __restrict__ aout) {              // [comp][b][h][S][128]
    __shared__ __align__(16) unsigned short Pls[8][16 * PKP];

    const int tid  = threadIdx.x;
    const int lane = tid & 31;
    const int w    = tid >> 5;
    const int half = lane >> 4;
    const int l16  = lane & 15;

    const int q0   = blockIdx.x * 128;
    const int bh   = blockIdx.y;           // b*H + h
    const int b    = bh / H_;
    const int h    = bh % H_;
    const int comp = blockIdx.z;
    const int hk   = h >> 1;               // NREP = 2

    const unsigned short* Q  = qb + (((size_t)(b * H_  + h ) * 2 + comp)) * S_ * D_;
    const unsigned short* Kp = kb + (((size_t)(b * HKV_ + hk) * 2 + comp)) * S_ * D_;
    const unsigned short* Vp = vT + ((size_t)(b * HKV_ + hk)) * DV_ * S_;
    float* Ao = aout + ((size_t)comp * B_ * H_ + bh) * (size_t)S_ * DV_;

    const int qr = q0 + w * 16;            // this wave's first query row

    v16bf qf[2];
    qf[0] = frag16(&Q[(size_t)(qr + l16) * D_ + 0],  half);
    qf[1] = frag16(&Q[(size_t)(qr + l16) * D_ + 32], half);

    v8f acc[8] = {};
    float mi_[8], li_[8];
#pragma unroll
    for (int r = 0; r < 8; ++r) { mi_[r] = -3e38f; li_[r] = 0.0f; }

    const int tEnd = (qr + 15) / 32;       // inclusive
    for (int t = 0; t <= tEnd; ++t) {
        const int k0 = t * 32;

        // S = Q * K^T  (16 x 32 scores)
        v8f sc[2] = {};
#pragma unroll
        for (int nt = 0; nt < 2; ++nt)
#pragma unroll
            for (int ks = 0; ks < 2; ++ks) {
                v16bf kf = frag16(&Kp[(size_t)(k0 + nt * 16 + l16) * D_ + ks * 32], half);
                sc[nt] = wmma_bf16(qf[ks], kf, sc[nt]);
            }

        // Issue V-fragment loads NOW: their global latency hides behind the
        // whole softmax VALU chain and the P->LDS staging below.
        v16bf vf[8];
#pragma unroll
        for (int dt = 0; dt < 8; ++dt)
            vf[dt] = frag16(&Vp[(size_t)(dt * 16 + l16) * S_ + k0], half);

        // mask + online softmax
        float p0v[8], p1v[8];
#pragma unroll
        for (int r = 0; r < 8; ++r) {
            int qrow = qr + r + half * 8;
            float s0 = sc[0][r] * SCALE_; if (k0 + l16 > qrow)      s0 = NEGINF_;
            float s1 = sc[1][r] * SCALE_; if (k0 + 16 + l16 > qrow) s1 = NEGINF_;
            float tmax = fmaxf(s0, s1);
#pragma unroll
            for (int off = 1; off < 16; off <<= 1)
                tmax = fmaxf(tmax, __shfl_xor(tmax, off, 16));
            float nm    = fmaxf(mi_[r], tmax);
            float alpha = __expf(mi_[r] - nm);
            float p0 = __expf(s0 - nm);
            float p1 = __expf(s1 - nm);
            float rs = p0 + p1;
#pragma unroll
            for (int off = 1; off < 16; off <<= 1)
                rs += __shfl_xor(rs, off, 16);
            li_[r] = li_[r] * alpha + rs;
            mi_[r] = nm;
            p0v[r] = p0; p1v[r] = p1;
#pragma unroll
            for (int dt = 0; dt < 8; ++dt)
                acc[dt][r] = acc[dt][r] * alpha;
        }

        // P -> per-wave LDS (bf16), then A-fragment (same-wave LDS is in-order)
        unsigned short* Pw = Pls[w];
#pragma unroll
        for (int r = 0; r < 8; ++r) {
            Pw[(r + half * 8) * PKP + l16]      = f2bf(p0v[r]);
            Pw[(r + half * 8) * PKP + 16 + l16] = f2bf(p1v[r]);
        }
        v16bf pf = frag16(&Pw[l16 * PKP], half);

        // acc += P * V
#pragma unroll
        for (int dt = 0; dt < 8; ++dt)
            acc[dt] = wmma_bf16(pf, vf[dt], acc[dt]);
    }

#pragma unroll
    for (int r = 0; r < 8; ++r) {
        float inv = 1.0f / li_[r];
#pragma unroll
        for (int dt = 0; dt < 8; ++dt) {
            int row = qr + r + half * 8;
            Ao[(size_t)row * DV_ + dt * 16 + l16] = acc[dt][r] * inv;
        }
    }
}

// ---------------- combine: a1 - lam*a2, RMS-norm, subln, (1-l0), -> bf16 ----------
__global__ __launch_bounds__(128)
void combine_kernel(const float* __restrict__ a1, const float* __restrict__ a2,
                    const float* __restrict__ lamp, const float* __restrict__ subw,
                    unsigned short* __restrict__ out, float one_minus_l0) {
    int idx = blockIdx.x;                 // (b*S + s)*H + h
    int h   = idx % H_;
    int bs  = idx / H_;
    int b   = bs / S_;
    int s   = bs % S_;
    int dv  = threadIdx.x;
    size_t arow = ((size_t)(b * H_ + h) * S_ + s) * DV_;
    float lam = lamp[0];
    float v = a1[arow + dv] - lam * a2[arow + dv];
    float sq = v * v;
#pragma unroll
    for (int off = 1; off < 32; off <<= 1) sq += __shfl_xor(sq, off, 32);
    __shared__ float red[4];
    if ((threadIdx.x & 31) == 0) red[threadIdx.x >> 5] = sq;
    __syncthreads();
    float tot = red[0] + red[1] + red[2] + red[3];
    float r = rsqrtf(tot * (1.0f / 128.0f) + 1e-5f);
    out[(size_t)bs * E_ + h * DV_ + dv] = f2bf(v * r * subw[dv] * one_minus_l0);
}

// ---------------- host launch ----------------
extern "C" void kernel_launch(void* const* d_in, const int* in_sizes, int n_in,
                              void* d_out, int out_size, void* d_ws, size_t ws_size,
                              hipStream_t stream) {
    const float* x    = (const float*)d_in[0];
    const float* cosT = (const float*)d_in[1];
    const float* sinT = (const float*)d_in[2];
    const float* Wq   = (const float*)d_in[3];
    const float* Wk   = (const float*)d_in[4];
    const float* Wv   = (const float*)d_in[5];
    const float* Wo   = (const float*)d_in[6];
    const float* lq1  = (const float*)d_in[7];
    const float* lk1  = (const float*)d_in[8];
    const float* lq2  = (const float*)d_in[9];
    const float* lk2  = (const float*)d_in[10];
    const float* subw = (const float*)d_in[11];
    float* out = (float*)d_out;

    const size_t NX   = (size_t)B_ * S_ * E_;        // 8,388,608
    const size_t NWQ  = (size_t)E_ * E_;             // 4,194,304
    const size_t NWK  = (size_t)(E_ / 2) * E_;       // 2,097,152
    const size_t M    = (size_t)B_ * S_;             // 4096

    char* w = (char*)d_ws;
    size_t o = 0;
    auto alloc = [&](size_t bytes) { size_t r = o; o += (bytes + 255) & ~(size_t)255; return r; };

    unsigned short* xb   = (unsigned short*)(w + alloc(NX * 2));
    unsigned short* wqb  = (unsigned short*)(w + alloc(NWQ * 2));
    unsigned short* wkb  = (unsigned short*)(w + alloc(NWK * 2));
    unsigned short* wvb  = (unsigned short*)(w + alloc(NWK * 2));
    unsigned short* wob  = (unsigned short*)(w + alloc(NWQ * 2));
    float*          Qf   = (float*)(w + alloc(M * 2048 * 4));
    float*          Kf   = (float*)(w + alloc(M * 1024 * 4));
    float*          Vf   = (float*)(w + alloc(M * 1024 * 4));
    unsigned short* qb   = (unsigned short*)(w + alloc(M * 2048 * 2));
    unsigned short* kb   = (unsigned short*)(w + alloc(M * 1024 * 2));
    unsigned short* vTb  = (unsigned short*)(w + alloc(M * 1024 * 2));
    float*          a12  = (float*)(w + alloc((size_t)2 * B_ * H_ * S_ * DV_ * 4));
    unsigned short* attb = (unsigned short*)(w + alloc(NX * 2));
    float*          lamp = (float*)(w + alloc(256));

    const float l0 = 0.8f - 0.6f * expf(-0.3f * 12.0f);

    // 1) bf16 conversions
    cvt_kernel<<<(NX  + 255) / 256, 256, 0, stream>>>(x,  xb,  (int)NX);
    cvt_kernel<<<(NWQ + 255) / 256, 256, 0, stream>>>(Wq, wqb, (int)NWQ);
    cvt_kernel<<<(NWK + 255) / 256, 256, 0, stream>>>(Wk, wkb, (int)NWK);
    cvt_kernel<<<(NWK + 255) / 256, 256, 0, stream>>>(Wv, wvb, (int)NWK);
    cvt_kernel<<<(NWQ + 255) / 256, 256, 0, stream>>>(Wo, wob, (int)NWQ);

    // 2) projections (WMMA GEMMs, TDM-fed)
    gemm_bf16_kernel<<<dim3(2048 / BN, M / BM), 256, 0, stream>>>(xb, wqb, Qf, (int)M, 2048, 2048);
    gemm_bf16_kernel<<<dim3(1024 / BN, M / BM), 256, 0, stream>>>(xb, wkb, Kf, (int)M, 1024, 2048);
    gemm_bf16_kernel<<<dim3(1024 / BN, M / BM), 256, 0, stream>>>(xb, wvb, Vf, (int)M, 1024, 2048);

    // 3) rotary + relayout (Q, K) and V transpose
    int rq = (int)(M * 2 * H_  * 32);
    int rk = (int)(M * 2 * HKV_ * 32);
    rope_kernel<<<(rq + 255) / 256, 256, 0, stream>>>(Qf, cosT, sinT, qb, 2 * H_,  rq);
    rope_kernel<<<(rk + 255) / 256, 256, 0, stream>>>(Kf, cosT, sinT, kb, 2 * HKV_, rk);
    int nv = (int)(M * 1024);
    vtrans_kernel<<<(nv + 255) / 256, 256, 0, stream>>>(Vf, vTb, nv);

    // 4) lambda scalar
    lam_kernel<<<1, 64, 0, stream>>>(lq1, lk1, lq2, lk2, l0, lamp);

    // 5) dual flash attention (comp = 0,1 via grid.z)
    attn_kernel<<<dim3(S_ / 128, B_ * H_, 2), 256, 0, stream>>>(qb, kb, vTb, a12);

    // 6) combine + RMS-norm -> bf16
    float* a1 = a12;
    float* a2 = a12 + (size_t)B_ * H_ * S_ * DV_;
    combine_kernel<<<(unsigned)(M * H_), 128, 0, stream>>>(a1, a2, lamp, subw, attb, 1.0f - l0);

    // 7) output projection (WMMA GEMM) -> f32 d_out
    gemm_bf16_kernel<<<dim3(2048 / BN, M / BM), 256, 0, stream>>>(attb, wob, out, (int)M, 2048, 2048);
}
